// CustomLSTM_12000138625181
// MI455X (gfx1250) — matrix operand order repaired and checked by reading, not compiled, not run on hardware
//
#include <hip/hip_runtime.h>
#include <math.h>

// ---------------------------------------------------------------------------
// LSTM on MI455X (gfx1250, wave32, WMMA bf16 16x16x32)
//
// gates[b, n] = sum_k [x_t | h]_{b,k} * W_{k,n} + bias_n     (K = 1024)
// W column layout (built by lstm_build_W): per workgroup wg, 32 columns
//   col = gate*8 + dk,  actual output unit kout = wg*8 + dk, gate in {i,f,c,o}
// stored K-contiguous: W[wg][col][k], so a B-fragment per lane is two
// 16-byte runs, 16 bytes apart, in LDS (two ds_load_b128).
//
// The recurrence is latency-bound (2048 sequential steps, ~134 MFLOP each),
// so everything data-parallel (weight repack, x fp32->bf16) is hoisted into
// one-shot kernels and the persistent kernel keeps:
//   - the 64 KB fused [U;V] panel in LDS for all 2048 steps,
//   - the cell state c in registers (2 floats/thread),
//   - h exchanged through a 32 KB double-buffered bf16 global (L2-resident),
//   - one atomic-counter grid barrier per step.
// ---------------------------------------------------------------------------

typedef __attribute__((ext_vector_type(16))) __bf16 v16bf;
typedef __attribute__((ext_vector_type(8)))  float  v8f;

#define BATCH 32
#define SEQ   2048
#define HID   512
#define KTOT  1024          // [x | h]
#define NWG   64            // persistent workgroups (grid barrier members)
#define WGTHREADS 128       // 4 waves: (Mtile, Ntile) = (w&1, w>>1)
#define NCOLS 32            // gate-interleaved columns per WG
#define W_ELEMS ((size_t)NWG * NCOLS * KTOT)          // 2,097,152 bf16 = 4 MB
#define HB_ELEMS (2 * BATCH * HID)                    // double-buffered h (bf16)
#define XB_ELEMS ((size_t)SEQ * BATCH * HID)          // bf16 x, [T][B][H] = 64 MB

// ---------------------------------------------------------------------------
// Zero h double-buffer and the grid-barrier counter (re-run every launch so
// the persistent kernel is deterministic under graph replay).
// ---------------------------------------------------------------------------
__global__ void lstm_init(unsigned short* hb, int* counter) {
    int idx = blockIdx.x * blockDim.x + threadIdx.x;
    if (idx < HB_ELEMS) hb[idx] = 0;          // bf16 +0.0
    if (idx == 0) *counter = 0;
}

// ---------------------------------------------------------------------------
// Build the fused, pre-permuted bf16 weight matrix W[wg][col][k].
//   k <  512 : U_gate[k,   kout]   (input projection)
//   k >= 512 : V_gate[k-512, kout] (recurrent projection)
// ---------------------------------------------------------------------------
__global__ void lstm_build_W(const float* __restrict__ Ui, const float* __restrict__ Uf,
                             const float* __restrict__ Uc, const float* __restrict__ Uo,
                             const float* __restrict__ Vi, const float* __restrict__ Vf,
                             const float* __restrict__ Vc, const float* __restrict__ Vo,
                             __bf16* __restrict__ W) {
    size_t idx = (size_t)blockIdx.x * blockDim.x + threadIdx.x;
    if (idx >= W_ELEMS) return;
    int k    = (int)(idx & (KTOT - 1));
    int col  = (int)((idx >> 10) & (NCOLS - 1));
    int wg   = (int)(idx >> 15);
    int gate = col >> 3;
    int kout = wg * 8 + (col & 7);
    const float* src;
    if (k < HID) {
        src = (gate == 0 ? Ui : gate == 1 ? Uf : gate == 2 ? Uc : Uo) + (size_t)k * HID + kout;
    } else {
        src = (gate == 0 ? Vi : gate == 1 ? Vf : gate == 2 ? Vc : Vo) + (size_t)(k - HID) * HID + kout;
    }
    W[idx] = (__bf16)(*src);
}

// ---------------------------------------------------------------------------
// One-shot x conversion/transpose: x[B][T][H] fp32 -> xb[T][B][H] bf16.
// Fully parallel, HBM-bandwidth-bound (~8 us); removes all v_cvt_pk_bf16_f32
// work from the sequential recurrence.
// ---------------------------------------------------------------------------
__global__ void lstm_convert_x(const float* __restrict__ x, __bf16* __restrict__ xb) {
    size_t idx = (size_t)blockIdx.x * blockDim.x + threadIdx.x;
    if (idx >= XB_ELEMS) return;
    int h = (int)(idx & (HID - 1));
    int b = (int)((idx >> 9) & (BATCH - 1));
    size_t t = idx >> 14;
    xb[idx] = (__bf16)x[((size_t)b * SEQ + t) * HID + h];
}

__device__ __forceinline__ float sigmoidf_(float v) {
    return 1.0f / (1.0f + __expf(-v));
}

// ---------------------------------------------------------------------------
// Persistent recurrence kernel. 64 WGs x 128 threads, grid-synced each step.
// LDS: 64 KB weight panel (loaded once) + 4 KB gate staging buffer.
// XB=true: x already bf16 in [T][B][H]; XB=false: convert fp32 on the fly.
// ---------------------------------------------------------------------------
template <bool XB>
__global__ void __launch_bounds__(WGTHREADS, 1)
lstm_persistent(const float* __restrict__ x,
                const __bf16* __restrict__ xb,
                const __bf16* __restrict__ W,
                const float* __restrict__ bi, const float* __restrict__ bf,
                const float* __restrict__ bc, const float* __restrict__ bo,
                unsigned short* __restrict__ hb,   // [2][BATCH][HID] bf16
                int* __restrict__ counter,
                float* __restrict__ out) {         // hidden_seq | h_T | c_T
    extern __shared__ char smem[];
    unsigned short* Wlds = (unsigned short*)smem;             // [NCOLS][KTOT] 64 KB
    float*          glds = (float*)(smem + NCOLS * KTOT * 2); // [BATCH][NCOLS] 4 KB

    const int wg    = blockIdx.x;
    const int tid   = threadIdx.x;
    const int lane  = tid & 31;
    const int wave  = tid >> 5;
    const int Mtile = wave & 1;        // rows 0-15 / 16-31 of batch
    const int Ntile = wave >> 1;       // 16-column group
    const int lhalf = (lane < 16) ? 0 : 1;
    const int koffA = lhalf ? 8 : 0;   // A-frag K sub-offset per documented layout
    const int kloB  = lhalf ? 8 : 0;   // B-frag K sub-offset per documented layout
    const int brow  = Mtile * 16 + (lane & 15);      // batch row this lane owns in A/C
    const int col_local = Ntile * 16 + (lane & 15);  // column this lane owns in B/C

    // ---- stage this WG's 64 KB weight panel into LDS (ds_store_b128) ----
    {
        const uint4* src = (const uint4*)(W + (size_t)wg * NCOLS * KTOT);
        uint4* dst = (uint4*)Wlds;
        const int n128 = NCOLS * KTOT / 8;   // 4096 x 16B
        for (int i = tid; i < n128; i += WGTHREADS) dst[i] = src[i];
    }

    // ---- per-lane bias, folded into the WMMA accumulator init ----
    const int   gsel = col_local >> 3;
    const int   kout_lane = wg * 8 + (col_local & 7);
    const float biasv = (gsel == 0 ? bi : gsel == 1 ? bf : gsel == 2 ? bc : bo)[kout_lane];
    v8f accb;
#pragma unroll
    for (int j = 0; j < 8; ++j) accb[j] = biasv;

    float creg[2] = {0.0f, 0.0f};   // cell state lives in registers all 2048 steps

    float* outHT = out + (size_t)BATCH * SEQ * HID;
    float* outCT = outHT + BATCH * HID;

    __syncthreads();

    for (int t = 0; t < SEQ; ++t) {
        const unsigned short* hbcur = hb + ((t & 1) ? BATCH * HID : 0);
        unsigned short*       hbnxt = hb + ((t & 1) ? 0 : BATCH * HID);

        v8f acc = accb;
        const float*  xrow  = x  + ((size_t)brow * SEQ + t) * HID;        // [B][T][H]
        const __bf16* xbrow = xb + ((size_t)t * BATCH + brow) * HID;      // [T][B][H]
        const unsigned short* hrow = hbcur + brow * HID;
        const unsigned short* wcol = Wlds + (size_t)col_local * KTOT;

        // ---- K = 0..511 : x_t contribution ----
#pragma unroll 4
        for (int kc = 0; kc < HID; kc += 32) {
            union { __bf16 e[16]; uint4 u[2]; v16bf v; } A;
            if (XB) {
                const uint4* xp = (const uint4*)(xbrow + kc + koffA);
                A.u[0] = xp[0];      // K = kc+koffA    .. +7
                A.u[1] = xp[2];      // K = kc+koffA+16 .. +23
            } else {
                const float* p = xrow + kc + koffA;
                float4 f0 = *(const float4*)(p);
                float4 f1 = *(const float4*)(p + 4);
                float4 f2 = *(const float4*)(p + 16);
                float4 f3 = *(const float4*)(p + 20);
                A.e[0]=(__bf16)f0.x; A.e[1]=(__bf16)f0.y; A.e[2]=(__bf16)f0.z; A.e[3]=(__bf16)f0.w;
                A.e[4]=(__bf16)f1.x; A.e[5]=(__bf16)f1.y; A.e[6]=(__bf16)f1.z; A.e[7]=(__bf16)f1.w;
                A.e[8]=(__bf16)f2.x; A.e[9]=(__bf16)f2.y; A.e[10]=(__bf16)f2.z; A.e[11]=(__bf16)f2.w;
                A.e[12]=(__bf16)f3.x; A.e[13]=(__bf16)f3.y; A.e[14]=(__bf16)f3.z; A.e[15]=(__bf16)f3.w;
            }

            union { uint4 u[2]; v16bf v; } B;
            const uint4* wp = (const uint4*)(wcol + kc + kloB);
            B.u[0] = wp[0]; B.u[1] = wp[2];

            acc = __builtin_amdgcn_wmma_f32_16x16x32_bf16(
                false, A.v, false, B.v, (short)0, acc, false, false);
        }

        // ---- K = 512..1023 : h_{t-1} contribution (bf16 straight loads) ----
#pragma unroll 4
        for (int kc = 0; kc < HID; kc += 32) {
            union { uint4 u[2]; v16bf v; } A;
            const uint4* hp = (const uint4*)(hrow + kc + koffA);
            A.u[0] = hp[0];
            A.u[1] = hp[2];

            union { uint4 u[2]; v16bf v; } B;
            const uint4* wp = (const uint4*)(wcol + HID + kc + kloB);
            B.u[0] = wp[0]; B.u[1] = wp[2];

            acc = __builtin_amdgcn_wmma_f32_16x16x32_bf16(
                false, A.v, false, B.v, (short)0, acc, false, false);
        }

        // ---- scatter C tile to LDS per documented C layout ----
#pragma unroll
        for (int j = 0; j < 8; ++j) {
            int row = Mtile * 16 + j + (lhalf ? 8 : 0);
            glds[row * NCOLS + col_local] = acc[j];
        }
        __syncthreads();

        // ---- elementwise gates: 256 (b, k) cells, 2 per thread ----
#pragma unroll
        for (int r = 0; r < 2; ++r) {
            int e  = tid + r * WGTHREADS;
            int b  = e >> 3;
            int dk = e & 7;
            float ip = glds[b * NCOLS + dk];
            float fp = glds[b * NCOLS + 8 + dk];
            float gp = glds[b * NCOLS + 16 + dk];
            float op = glds[b * NCOLS + 24 + dk];
            float it = sigmoidf_(ip);
            float ft = sigmoidf_(fp);
            float gt = tanhf(gp);
            float ot = sigmoidf_(op);
            float cn = ft * creg[r] + it * gt;
            float hn = ot * tanhf(cn);
            creg[r] = cn;
            int kout = wg * 8 + dk;
            out[((size_t)b * SEQ + t) * HID + kout] = hn;           // hidden_seq [B,T,H]
            ((__bf16*)hbnxt)[b * HID + kout] = (__bf16)hn;          // next-step GEMM input
            if (t == SEQ - 1) {
                outHT[b * HID + kout] = hn;
                outCT[b * HID + kout] = cn;
            }
        }

        // ---- grid barrier: h_{t} must be device-visible before step t+1 ----
        __threadfence();
        __syncthreads();
        if (tid == 0) {
            atomicAdd(counter, 1);
            const int target = NWG * (t + 1);
            volatile int* vcnt = counter;
            while (*vcnt < target) { __builtin_amdgcn_s_sleep(2); }
        }
        __syncthreads();   // also fences glds reuse next iteration
    }
}

extern "C" void kernel_launch(void* const* d_in, const int* in_sizes, int n_in,
                              void* d_out, int out_size, void* d_ws, size_t ws_size,
                              hipStream_t stream) {
    (void)in_sizes; (void)n_in; (void)out_size;
    const float* x  = (const float*)d_in[0];
    const float* Ui = (const float*)d_in[1];
    const float* Vi = (const float*)d_in[2];
    const float* bi = (const float*)d_in[3];
    const float* Uf = (const float*)d_in[4];
    const float* Vf = (const float*)d_in[5];
    const float* bf = (const float*)d_in[6];
    const float* Uc = (const float*)d_in[7];
    const float* Vc = (const float*)d_in[8];
    const float* bc = (const float*)d_in[9];
    const float* Uo = (const float*)d_in[10];
    const float* Vo = (const float*)d_in[11];
    const float* bo = (const float*)d_in[12];

    // workspace carve-up:
    //   | W (4 MB bf16) | h double-buffer (64 KB) | counter (pad 256B) | xb (64 MB, optional) |
    const size_t offW   = 0;
    const size_t offHB  = offW + W_ELEMS * 2;
    const size_t offCNT = offHB + HB_ELEMS * 2;
    const size_t offXB  = (offCNT + 4 + 255) & ~(size_t)255;
    const size_t needXB = offXB + XB_ELEMS * 2;

    __bf16*         W       = (__bf16*)((char*)d_ws + offW);
    unsigned short* hbuf    = (unsigned short*)((char*)d_ws + offHB);
    int*            counter = (int*)((char*)d_ws + offCNT);
    __bf16*         xbuf    = (__bf16*)((char*)d_ws + offXB);
    float*          out     = (float*)d_out;

    const bool useXB = (ws_size >= needXB);

    lstm_init<<<(HB_ELEMS + 255) / 256, 256, 0, stream>>>(hbuf, counter);
    lstm_build_W<<<(int)((W_ELEMS + 255) / 256), 256, 0, stream>>>(
        Ui, Uf, Uc, Uo, Vi, Vf, Vc, Vo, W);

    const size_t smem = (size_t)NCOLS * KTOT * 2 /*W panel*/ + (size_t)BATCH * NCOLS * 4 /*gates*/;
    if (useXB) {
        lstm_convert_x<<<(int)((XB_ELEMS + 255) / 256), 256, 0, stream>>>(x, xbuf);
        lstm_persistent<true><<<NWG, WGTHREADS, smem, stream>>>(
            x, xbuf, W, bi, bf, bc, bo, hbuf, counter, out);
    } else {
        lstm_persistent<false><<<NWG, WGTHREADS, smem, stream>>>(
            x, xbuf, W, bi, bf, bc, bo, hbuf, counter, out);
    }
}
